// SGMultiHeadAttention_34402688041236
// MI455X (gfx1250) — compile-verified, matrix operand order checked by reading, not run on hardware
//
#include <hip/hip_runtime.h>
#include <math.h>

#define S_LEN   1024
#define D_MODEL 1024
#define NHEAD   16
#define HEAD_DK 64
#define BATCH   8
#define SCALE_F 0.125f   // 1/sqrt(64)

typedef __bf16 bf16;
typedef __attribute__((ext_vector_type(16))) __bf16 v16bf;
typedef __attribute__((ext_vector_type(8)))  float  v8f;
typedef __attribute__((ext_vector_type(4)))  unsigned int u32x4;

// Load a 16-bit WMMA operand fragment (v16bf) from a row-major LDS tile.
// p points at (row*stride + kchunk_base + c0) where c0 = (lane>=16)*8.
// Two 16-byte chunks: elements [0..7] at p, elements [8..15] at p+16.
__device__ __forceinline__ v16bf ld_frag(const bf16* p) {
  union { v16bf v; u32x4 u[2]; } f;
  f.u[0] = *(const u32x4*)(p);
  f.u[1] = *(const u32x4*)(p + 16);
  return f.v;
}

__device__ __forceinline__ v8f wmma_bf16(v16bf a, v16bf b, v8f c) {
  return __builtin_amdgcn_wmma_f32_16x16x32_bf16(false, a, false, b,
                                                 (short)0, c, false, false);
}

// ======================================================================
// Prep 1: elementwise fp32 -> bf16 (coalesced float4 -> 4x bf16).
// ======================================================================
__global__ __launch_bounds__(256)
void cvt_f32_bf16_kernel(const float* __restrict__ in, bf16* __restrict__ out,
                         int n4)
{
  int i = blockIdx.x * 256 + threadIdx.x;
  if (i >= n4) return;
  float4 d = ((const float4*)in)[i];
  union { bf16 e[4]; uint2 u; } o;
  o.e[0] = (bf16)d.x; o.e[1] = (bf16)d.y;
  o.e[2] = (bf16)d.z; o.e[3] = (bf16)d.w;
  ((uint2*)out)[i] = o.u;
}

// ======================================================================
// Prep 2: W[K,N] fp32 -> Wt[N,K] bf16 (tiled 64x64 transpose via LDS).
// ======================================================================
__global__ __launch_bounds__(256)
void wt_cvt_kernel(const float* __restrict__ W, bf16* __restrict__ Wt,
                   int K, int N)
{
  __shared__ __align__(16) bf16 sT[64 * 72];
  const int kb = blockIdx.x * 64, nb = blockIdx.y * 64;
  const int tid = threadIdx.x;
#pragma unroll
  for (int i = 0; i < 4; ++i) {                 // read coalesced along N
    int f4 = tid + i * 256;
    int r = f4 >> 4, c4 = (f4 & 15) * 4;
    float4 d = *(const float4*)(W + (size_t)(kb + r) * N + nb + c4);
    sT[(c4 + 0) * 72 + r] = (bf16)d.x;
    sT[(c4 + 1) * 72 + r] = (bf16)d.y;
    sT[(c4 + 2) * 72 + r] = (bf16)d.z;
    sT[(c4 + 3) * 72 + r] = (bf16)d.w;
  }
  __syncthreads();
#pragma unroll
  for (int i = 0; i < 2; ++i) {                 // write coalesced along K
    int f8 = tid + i * 256;
    int n = f8 >> 3, k8 = (f8 & 7) * 8;
    *(u32x4*)(Wt + (size_t)(nb + n) * K + kb + k8) =
        *(const u32x4*)(sT + n * 72 + k8);
  }
}

// ======================================================================
// GEMM: Out[M,N] = A[M,K] @ Wt[N,K]^T + bias.  BM=BN=128, BK=64, 8 waves.
// A and Wt are bf16; staging is pure b128 copies (no conversions).
// OUT_F32: fp32 row-major [M,N]; else bf16 head-split [B,H,S,64].
// ======================================================================
template <int OUT_F32>
__global__ __launch_bounds__(256)
void gemm_bf16_kernel(const bf16* __restrict__ A, const bf16* __restrict__ Wt,
                      const float* __restrict__ bias, void* __restrict__ Out,
                      int M, int K, int N)
{
  __shared__ __align__(16) bf16 sA[128 * 72];   // [m][k], pad 8
  __shared__ __align__(16) bf16 sB[128 * 72];   // [n][k], pad 8
  const int tid   = threadIdx.x;
  const int lane  = tid & 31;
  const int wave  = tid >> 5;
  const int wr    = wave & 3;          // 4x2 wave grid: 32 rows x 64 cols each
  const int wc    = wave >> 2;
  const int lcol  = lane & 15;
  const int lhalf = lane >> 4;
  const int c0    = lhalf * 8;
  const int rowb  = blockIdx.y * 128;
  const int colb  = blockIdx.x * 128;

  v8f acc[2][4] = {};

  for (int kb = 0; kb < K; kb += 64) {
#pragma unroll
    for (int i = 0; i < 4; ++i) {               // 1024 chunks of 8 bf16 each
      int f8 = tid + i * 256;
      int m = f8 >> 3, k8 = (f8 & 7) * 8;
      *(u32x4*)(sA + m * 72 + k8) =
          *(const u32x4*)(A + (size_t)(rowb + m) * K + kb + k8);
      *(u32x4*)(sB + m * 72 + k8) =
          *(const u32x4*)(Wt + (size_t)(colb + m) * K + kb + k8);
    }
    __syncthreads();
#pragma unroll
    for (int j = 0; j < 2; ++j) {               // two k-chunks of 32
      v16bf aF[2], bF[4];
#pragma unroll
      for (int tm = 0; tm < 2; ++tm)
        aF[tm] = ld_frag(sA + (wr * 32 + tm * 16 + lcol) * 72 + j * 32 + c0);
#pragma unroll
      for (int tn = 0; tn < 4; ++tn)
        bF[tn] = ld_frag(sB + (wc * 64 + tn * 16 + lcol) * 72 + j * 32 + c0);
#pragma unroll
      for (int tm = 0; tm < 2; ++tm)
#pragma unroll
        for (int tn = 0; tn < 4; ++tn)
          acc[tm][tn] = wmma_bf16(aF[tm], bF[tn], acc[tm][tn]);
    }
    __syncthreads();
  }

#pragma unroll
  for (int tm = 0; tm < 2; ++tm) {
#pragma unroll
    for (int tn = 0; tn < 4; ++tn) {
      int col = colb + wc * 64 + tn * 16 + lcol;
      float bv = bias[col];
#pragma unroll
      for (int r = 0; r < 8; ++r) {
        int row = rowb + wr * 32 + tm * 16 + r + lhalf * 8;
        float val = acc[tm][tn][r] + bv;
        if (OUT_F32) {
          ((float*)Out)[(size_t)row * N + col] = val;
        } else {
          int b = row >> 10, s = row & (S_LEN - 1);
          int h = col >> 6,  dk = col & (HEAD_DK - 1);
          ((bf16*)Out)[(((size_t)(b * NHEAD + h) * S_LEN) + s) * HEAD_DK + dk] =
              (bf16)val;
        }
      }
    }
  }
}

// ======================================================================
// Pass 1: per-column softmax stats. scores^T = K @ Q^T (rows = key index),
// online row-wise (max, sumexp) with mask, butterfly-reduce across lanes.
// grid: (S/128 key-blocks, B*H). 8 waves x 16 key-rows.
// ======================================================================
__global__ __launch_bounds__(256)
void attn_stats_kernel(const bf16* __restrict__ Kh, const bf16* __restrict__ Qh,
                       const float* __restrict__ mask,
                       float* __restrict__ mstat, float* __restrict__ zinv)
{
  __shared__ __align__(16) bf16 sK[128 * 72];  // [krow][feat], pad 8
  __shared__ __align__(16) bf16 sQ[16 * 72];   // [q][feat]
  const int tid = threadIdx.x, lane = tid & 31, wave = tid >> 5;
  const int lcol = lane & 15, lhalf = lane >> 4, c0 = lhalf * 8;
  const int bh = blockIdx.y, b = bh >> 4;
  const int kbase = blockIdx.x * 128;
  const bf16* Kp = Kh + (size_t)bh * S_LEN * HEAD_DK;
  const bf16* Qp = Qh + (size_t)bh * S_LEN * HEAD_DK;

#pragma unroll
  for (int i = 0; i < 4; ++i) {                // stage 128x64 K block once
    int f8 = tid + i * 256;
    int r = f8 >> 3, c8 = (f8 & 7) * 8;
    *(u32x4*)(sK + r * 72 + c8) =
        *(const u32x4*)(Kp + (size_t)(kbase + r) * HEAD_DK + c8);
  }
  __syncthreads();

  v16bf aF[2];
#pragma unroll
  for (int j = 0; j < 2; ++j)
    aF[j] = ld_frag(sK + (wave * 16 + lcol) * 72 + j * 32 + c0);

  float mrow[8], zrow[8];
#pragma unroll
  for (int r = 0; r < 8; ++r) { mrow[r] = -3.0e38f; zrow[r] = 0.f; }

  for (int qt = 0; qt < S_LEN / 16; ++qt) {
    if (tid < 128) {                           // stage 16x64 Q tile
      int r = tid >> 3, c8 = (tid & 7) * 8;
      *(u32x4*)(sQ + r * 72 + c8) =
          *(const u32x4*)(Qp + (size_t)(qt * 16 + r) * HEAD_DK + c8);
    }
    __syncthreads();
    v16bf bF0 = ld_frag(sQ + lcol * 72 + c0);
    v16bf bF1 = ld_frag(sQ + lcol * 72 + 32 + c0);
    v8f a = {};
    a = wmma_bf16(aF[0], bF0, a);
    a = wmma_bf16(aF[1], bF1, a);

    int qcol = qt * 16 + lcol;
    const float* mp = mask + ((size_t)b * S_LEN + qcol) * S_LEN +
                      kbase + wave * 16 + lhalf * 8;
    float4 m0 = *(const float4*)(mp);
    float4 m1 = *(const float4*)(mp + 4);
    float mv[8] = {m0.x, m0.y, m0.z, m0.w, m1.x, m1.y, m1.z, m1.w};
#pragma unroll
    for (int r = 0; r < 8; ++r) {
      float s = a[r] * SCALE_F;
      s = (mv[r] == 0.f) ? -1.0e9f : s;
      float nm = fmaxf(mrow[r], s);
      zrow[r] = zrow[r] * __expf(mrow[r] - nm) + __expf(s - nm);
      mrow[r] = nm;
    }
    __syncthreads();
  }

  for (int d = 1; d < 16; d <<= 1) {           // reduce across q-lanes
#pragma unroll
    for (int r = 0; r < 8; ++r) {
      float om = __shfl_xor(mrow[r], d, 32);
      float oz = __shfl_xor(zrow[r], d, 32);
      float nm = fmaxf(mrow[r], om);
      zrow[r] = zrow[r] * __expf(mrow[r] - nm) + oz * __expf(om - nm);
      mrow[r] = nm;
    }
  }
  if (lcol == 0) {
#pragma unroll
    for (int r = 0; r < 8; ++r) {
      int krow = kbase + wave * 16 + r + lhalf * 8;
      mstat[(size_t)bh * S_LEN + krow] = mrow[r];
      zinv[(size_t)bh * S_LEN + krow]  = 1.0f / zrow[r];
    }
  }
}

// ======================================================================
// Pass 2: out[q,d] = sum_k exp(s[q,k]-m_k) * (V[k,d]/Z_k).
// grid: (S/128 q-blocks, B*H). Wave w owns q-rows [w*16,w*16+16), all 64 d.
// ======================================================================
__global__ __launch_bounds__(256)
void attn_out_kernel(const bf16* __restrict__ Qh, const bf16* __restrict__ Kh,
                     const bf16* __restrict__ Vh, const float* __restrict__ mask,
                     const float* __restrict__ mstat,
                     const float* __restrict__ zinv, bf16* __restrict__ ctx)
{
  __shared__ __align__(16) bf16 sQ[128 * 72];      // [q][feat]
  __shared__ __align__(16) bf16 sK[32 * 72];       // [k][feat]
  __shared__ __align__(16) bf16 sVt[64 * 40];      // [d][k], pre-scaled by 1/Z
  __shared__ __align__(16) bf16 sP[8][16 * 40];    // per-wave P tile [q][k]
  const int tid = threadIdx.x, lane = tid & 31, wave = tid >> 5;
  const int lcol = lane & 15, lhalf = lane >> 4, c0 = lhalf * 8;
  const int bh = blockIdx.y, b = bh >> 4, h = bh & 15;
  const int qbase = blockIdx.x * 128;
  const bf16* Qp = Qh + (size_t)bh * S_LEN * HEAD_DK;
  const bf16* Kp = Kh + (size_t)bh * S_LEN * HEAD_DK;
  const bf16* Vp = Vh + (size_t)bh * S_LEN * HEAD_DK;

#pragma unroll
  for (int i = 0; i < 4; ++i) {                    // stage 128x64 Q block once
    int f8 = tid + i * 256;
    int r = f8 >> 3, c8 = (f8 & 7) * 8;
    *(u32x4*)(sQ + r * 72 + c8) =
        *(const u32x4*)(Qp + (size_t)(qbase + r) * HEAD_DK + c8);
  }
  __syncthreads();
  v16bf qF[2];
#pragma unroll
  for (int j = 0; j < 2; ++j)
    qF[j] = ld_frag(sQ + (wave * 16 + lcol) * 72 + j * 32 + c0);

  v8f acco[4] = {};

  for (int kt = 0; kt < S_LEN; kt += 32) {
    {                                              // stage K tile + scaled V^T
      int r = tid >> 3, c8 = (tid & 7) * 8;        // 256 chunks = 32x64
      *(u32x4*)(sK + r * 72 + c8) =
          *(const u32x4*)(Kp + (size_t)(kt + r) * HEAD_DK + c8);
      float iz = zinv[(size_t)bh * S_LEN + kt + r];
      union { u32x4 u; bf16 e[8]; } vv;
      vv.u = *(const u32x4*)(Vp + (size_t)(kt + r) * HEAD_DK + c8);
#pragma unroll
      for (int j = 0; j < 8; ++j)
        sVt[(c8 + j) * 40 + r] = (bf16)((float)vv.e[j] * iz);
    }
    __syncthreads();

#pragma unroll
    for (int th = 0; th < 2; ++th) {               // two 16-col score tiles
      v16bf kF0 = ld_frag(sK + (th * 16 + lcol) * 72 + c0);
      v16bf kF1 = ld_frag(sK + (th * 16 + lcol) * 72 + 32 + c0);
      v8f a = {};
      a = wmma_bf16(qF[0], kF0, a);
      a = wmma_bf16(qF[1], kF1, a);
      int col = kt + th * 16 + lcol;
      float mc = mstat[(size_t)bh * S_LEN + col];
      const float* mrow = mask + (size_t)b * S_LEN * S_LEN + col;
#pragma unroll
      for (int r = 0; r < 8; ++r) {
        int q = qbase + wave * 16 + r + lhalf * 8;
        float s = a[r] * SCALE_F;
        float mv = mrow[(size_t)q * S_LEN];
        s = (mv == 0.f) ? -1.0e9f : s;
        float e = __expf(s - mc);
        sP[wave][(r + lhalf * 8) * 40 + th * 16 + lcol] = (bf16)e;
      }
    }
    v16bf pF = ld_frag(sP[wave] + lcol * 40 + c0); // P tile as A operand
#pragma unroll
    for (int dt = 0; dt < 4; ++dt) {
      v16bf vF = ld_frag(sVt + (dt * 16 + lcol) * 40 + c0);
      acco[dt] = wmma_bf16(pF, vF, acco[dt]);
    }
    __syncthreads();
  }

#pragma unroll
  for (int dt = 0; dt < 4; ++dt) {                 // ctx[b][q][h*64+d], bf16
#pragma unroll
    for (int r = 0; r < 8; ++r) {
      int q = qbase + wave * 16 + r + lhalf * 8;
      int d = dt * 16 + lcol;
      ctx[((size_t)b * S_LEN + q) * D_MODEL + h * HEAD_DK + d] = (bf16)acco[dt][r];
    }
  }
}

// ======================================================================
extern "C" void kernel_launch(void* const* d_in, const int* in_sizes, int n_in,
                              void* d_out, int out_size, void* d_ws,
                              size_t ws_size, hipStream_t stream) {
  (void)in_sizes; (void)n_in; (void)out_size; (void)ws_size;
  const float* q    = (const float*)d_in[0];
  const float* k    = (const float*)d_in[1];
  const float* v    = (const float*)d_in[2];
  const float* mask = (const float*)d_in[3];
  const float* Wq   = (const float*)d_in[4];
  const float* bq   = (const float*)d_in[5];
  const float* Wk   = (const float*)d_in[6];
  const float* bk   = (const float*)d_in[7];
  const float* Wv   = (const float*)d_in[8];
  const float* bv   = (const float*)d_in[9];
  const float* Wo   = (const float*)d_in[10];
  const float* bo   = (const float*)d_in[11];

  char* ws = (char*)d_ws;
  size_t off = 0;
  auto wsalloc = [&](size_t bytes) {
    void* p = ws + off;
    off += (bytes + 255) & ~(size_t)255;
    return p;
  };
  const size_t nBH   = (size_t)BATCH * NHEAD;
  const size_t nMD   = (size_t)BATCH * S_LEN * D_MODEL;       // 8.4M elems
  const size_t nWW   = (size_t)D_MODEL * D_MODEL;             // 1M elems
  bf16*  qb  = (bf16*)wsalloc(nMD * sizeof(bf16));
  bf16*  kb  = (bf16*)wsalloc(nMD * sizeof(bf16));
  bf16*  vb  = (bf16*)wsalloc(nMD * sizeof(bf16));
  bf16*  Wqt = (bf16*)wsalloc(nWW * sizeof(bf16));
  bf16*  Wkt = (bf16*)wsalloc(nWW * sizeof(bf16));
  bf16*  Wvt = (bf16*)wsalloc(nWW * sizeof(bf16));
  bf16*  Wot = (bf16*)wsalloc(nWW * sizeof(bf16));
  bf16*  Qh  = (bf16*)wsalloc(nBH * S_LEN * HEAD_DK * sizeof(bf16));
  bf16*  Kh  = (bf16*)wsalloc(nBH * S_LEN * HEAD_DK * sizeof(bf16));
  bf16*  Vh  = (bf16*)wsalloc(nBH * S_LEN * HEAD_DK * sizeof(bf16));
  float* ms  = (float*)wsalloc(nBH * S_LEN * sizeof(float));
  float* zi  = (float*)wsalloc(nBH * S_LEN * sizeof(float));
  bf16*  ctx = (bf16*)wsalloc(nMD * sizeof(bf16));

  const int M = BATCH * S_LEN;
  dim3 blk(256);
  dim3 gemm_grid(D_MODEL / 128, M / 128);
  dim3 attn_grid(S_LEN / 128, (unsigned)nBH);
  dim3 wt_grid(D_MODEL / 64, D_MODEL / 64);
  const int n4 = (int)(nMD / 4);
  dim3 cvt_grid((n4 + 255) / 256);

  // Prep: inputs -> bf16, weights -> transposed bf16.
  cvt_f32_bf16_kernel<<<cvt_grid, blk, 0, stream>>>(q, qb, n4);
  cvt_f32_bf16_kernel<<<cvt_grid, blk, 0, stream>>>(k, kb, n4);
  cvt_f32_bf16_kernel<<<cvt_grid, blk, 0, stream>>>(v, vb, n4);
  wt_cvt_kernel<<<wt_grid, blk, 0, stream>>>(Wq, Wqt, D_MODEL, D_MODEL);
  wt_cvt_kernel<<<wt_grid, blk, 0, stream>>>(Wk, Wkt, D_MODEL, D_MODEL);
  wt_cvt_kernel<<<wt_grid, blk, 0, stream>>>(Wv, Wvt, D_MODEL, D_MODEL);
  wt_cvt_kernel<<<wt_grid, blk, 0, stream>>>(Wo, Wot, D_MODEL, D_MODEL);

  // Projections -> head-split bf16 [B,H,S,64]
  gemm_bf16_kernel<0><<<gemm_grid, blk, 0, stream>>>(qb, Wqt, bq, Qh, M, D_MODEL, D_MODEL);
  gemm_bf16_kernel<0><<<gemm_grid, blk, 0, stream>>>(kb, Wkt, bk, Kh, M, D_MODEL, D_MODEL);
  gemm_bf16_kernel<0><<<gemm_grid, blk, 0, stream>>>(vb, Wvt, bv, Vh, M, D_MODEL, D_MODEL);
  // Column-softmax stats, then attention output (context, bf16 [B,S,D])
  attn_stats_kernel<<<attn_grid, blk, 0, stream>>>(Kh, Qh, mask, ms, zi);
  attn_out_kernel<<<attn_grid, blk, 0, stream>>>(Qh, Kh, Vh, mask, ms, zi, ctx);
  // Output projection -> fp32 [B,S,D]
  gemm_bf16_kernel<1><<<gemm_grid, blk, 0, stream>>>(ctx, Wot, bo, (float*)d_out, M, D_MODEL, D_MODEL);
}
